// LigerFusedLinearJSD_37967510896659
// MI455X (gfx1250) — compile-verified, matrix operand order checked by reading, not run on hardware
//
#include <hip/hip_runtime.h>

// Problem constants (match reference)
#define BT 2048
#define HDIM 2048
#define VOCAB 32000

// GEMM tiling
#define TILE_M 128
#define TILE_N 128
#define TILE_K 32
#define LDSPAD 40   // bf16 elements per LDS row (80 bytes: 16B aligned, conflict-free)

typedef __attribute__((ext_vector_type(16))) __bf16 v16bf;
typedef __attribute__((ext_vector_type(8)))  __bf16 v8bf;
typedef __attribute__((ext_vector_type(8)))  float  v8f;
typedef int v4i __attribute__((vector_size(16)));

// CDNA5 async global->LDS path (guarded: falls back to load+ds_store if absent)
#if defined(__has_builtin)
#  if __has_builtin(__builtin_amdgcn_global_load_async_to_lds_b128) && \
      __has_builtin(__builtin_amdgcn_s_wait_asynccnt)
#    define USE_ASYNC_LDS 1
#  endif
#endif
#ifndef USE_ASYNC_LDS
#  define USE_ASYNC_LDS 0
#endif

#if USE_ASYNC_LDS
// Builtin signature: (v4i AS1* src, v4i AS3* ldsDst, Imm offset, Imm cpol)
#  define TO_GLOBAL(p) ((__attribute__((address_space(1))) v4i*)(p))
#  define TO_LDS(p)    ((__attribute__((address_space(3))) v4i*)(p))
#  define ASYNC_WAIT() __builtin_amdgcn_s_wait_asynccnt(0)
#else
#  define ASYNC_WAIT()
#endif

__device__ __forceinline__ unsigned short f2bf(float f) {
    unsigned int u = __float_as_uint(f);
    unsigned int r = u + 0x7FFFu + ((u >> 16) & 1u);   // round-to-nearest-even
    return (unsigned short)(r >> 16);
}
__device__ __forceinline__ float bf2f(unsigned int h) {
    return __uint_as_float(h << 16);
}

__device__ __forceinline__ void atomicMaxF(float* addr, float val) {
    unsigned int* ua = (unsigned int*)addr;
    unsigned int old = *ua;
    while (true) {
        if (__uint_as_float(old) >= val) break;
        unsigned int assumed = old;
        old = atomicCAS(ua, assumed, __float_as_uint(val));
        if (old == assumed) break;
    }
}

// -------------------------------------------------------------------------
// K0: initialize row maxes to -inf, loss to 0
// -------------------------------------------------------------------------
__global__ void init_kernel(float* __restrict__ max_s, float* __restrict__ max_t,
                            float* __restrict__ out) {
    int i = blockIdx.x * blockDim.x + threadIdx.x;
    if (i < BT) {
        max_s[i] = -INFINITY;
        max_t[i] = -INFINITY;
    }
    if (i == 0) out[0] = 0.0f;
}

// -------------------------------------------------------------------------
// K0b: streaming fp32 -> bf16 conversion (memory bound; one pass, outside GEMM)
// -------------------------------------------------------------------------
__global__ __launch_bounds__(256) void cvt_bf16_kernel(
        const float* __restrict__ src, unsigned short* __restrict__ dst, int n4) {
    int i = blockIdx.x * blockDim.x + threadIdx.x;
    int stride = gridDim.x * blockDim.x;
    for (; i < n4; i += stride) {
        float4 v = ((const float4*)src)[i];
        ushort4 p;
        p.x = f2bf(v.x); p.y = f2bf(v.y); p.z = f2bf(v.z); p.w = f2bf(v.w);
        ((ushort4*)dst)[i] = p;
    }
}

// -------------------------------------------------------------------------
// K1: logits[bt, v] = A[bt, :] . B[v, :]   (bf16 operands, f32 WMMA accum)
// Block: 256 threads = 8 waves. Tile 128x128, K-step 32, double-buffered LDS
// staged via async global->LDS DMA. Wave grid 4(M) x 2(N): each wave 32x64.
// -------------------------------------------------------------------------
__global__ __launch_bounds__(256) void gemm_logits_kernel(
        const unsigned short* __restrict__ A,   // [BT, H] bf16 bits
        const unsigned short* __restrict__ B,   // [V, H]  bf16 bits
        unsigned short* __restrict__ logits,    // [BT, V] bf16 bits
        float* __restrict__ rowmax) {           // [BT]
    __shared__ __bf16 As[2][TILE_M * LDSPAD];
    __shared__ __bf16 Bs[2][TILE_N * LDSPAD];

    const int bn = blockIdx.x;       // vocab tile
    const int bm = blockIdx.y;       // token tile
    const int tid = threadIdx.x;
    const int lane = tid & 31;
    const int wid = tid >> 5;
    const int wave_m = (wid & 3) * 32;   // 0,32,64,96
    const int wave_n = (wid >> 2) * 64;  // 0,64
    const int half = lane >> 4;          // K-chunk selector per WMMA layout
    const int l16 = lane & 15;

    // Stage one 128x32 bf16 tile of A and of B into LDS (16B chunks).
    auto stage = [&](int buf, int k0) {
        #pragma unroll
        for (int t = 0; t < 2; ++t) {
            int idx = tid + t * 256;       // 0..511
            int row = idx >> 2;            // 0..127
            int q   = idx & 3;             // 16B chunk within the 64B row slice
            const unsigned short* ga = A + (size_t)(bm * TILE_M + row) * HDIM + k0 + q * 8;
            const unsigned short* gb = B + (size_t)(bn * TILE_N + row) * HDIM + k0 + q * 8;
            __bf16* la = &As[buf][row * LDSPAD + q * 8];
            __bf16* lb = &Bs[buf][row * LDSPAD + q * 8];
#if USE_ASYNC_LDS
            __builtin_amdgcn_global_load_async_to_lds_b128(TO_GLOBAL(ga), TO_LDS(la), 0, 0);
            __builtin_amdgcn_global_load_async_to_lds_b128(TO_GLOBAL(gb), TO_LDS(lb), 0, 0);
#else
            *(uint4*)la = *(const uint4*)ga;
            *(uint4*)lb = *(const uint4*)gb;
#endif
        }
    };

    v8f acc[2][4];
    #pragma unroll
    for (int i = 0; i < 2; ++i)
        #pragma unroll
        for (int j = 0; j < 4; ++j)
            #pragma unroll
            for (int e = 0; e < 8; ++e) acc[i][j][e] = 0.0f;

    stage(0, 0);
    ASYNC_WAIT();
    __syncthreads();

    const int KSTEPS = HDIM / TILE_K;   // 64
    for (int ks = 0; ks < KSTEPS; ++ks) {
        int cur = ks & 1;
        if (ks + 1 < KSTEPS) stage(cur ^ 1, (ks + 1) * TILE_K);

        // Fragments per 16-bit WMMA VGPR layout:
        // lanes 0-15 hold K 0..7 & 16..23; lanes 16-31 hold K 8..15 & 24..31.
        v16bf afrag[2];
        #pragma unroll
        for (int i = 0; i < 2; ++i) {
            const __bf16* base = &As[cur][(wave_m + i * 16 + l16) * LDSPAD];
            v8bf lo = *(const v8bf*)(base + half * 8);
            v8bf hi = *(const v8bf*)(base + 16 + half * 8);
            afrag[i] = __builtin_shufflevector(lo, hi,
                0, 1, 2, 3, 4, 5, 6, 7, 8, 9, 10, 11, 12, 13, 14, 15);
        }
        v16bf bfrag[4];
        #pragma unroll
        for (int j = 0; j < 4; ++j) {
            const __bf16* base = &Bs[cur][(wave_n + j * 16 + l16) * LDSPAD];
            v8bf lo = *(const v8bf*)(base + half * 8);
            v8bf hi = *(const v8bf*)(base + 16 + half * 8);
            bfrag[j] = __builtin_shufflevector(lo, hi,
                0, 1, 2, 3, 4, 5, 6, 7, 8, 9, 10, 11, 12, 13, 14, 15);
        }

        #pragma unroll
        for (int i = 0; i < 2; ++i)
            #pragma unroll
            for (int j = 0; j < 4; ++j)
                acc[i][j] = __builtin_amdgcn_wmma_f32_16x16x32_bf16(
                    false, afrag[i], false, bfrag[j],
                    (short)0, acc[i][j], false, false);

        ASYNC_WAIT();
        __syncthreads();
    }

    // Epilogue. C layout: lanes 0-15 -> N=l16, M=vg(0..7); lanes 16-31 -> M=8+vg.
    #pragma unroll
    for (int i = 0; i < 2; ++i) {
        int gr0 = bm * TILE_M + wave_m + i * 16 + half * 8;
        float mrow[8];
        #pragma unroll
        for (int vg = 0; vg < 8; ++vg) {
            int gr = gr0 + vg;
            float m = -INFINITY;
            #pragma unroll
            for (int j = 0; j < 4; ++j) {
                float v = acc[i][j][vg];   // inv_t == 1.0 folded out
                int gc = bn * TILE_N + wave_n + j * 16 + l16;
                logits[(size_t)gr * VOCAB + gc] = f2bf(v);
                m = fmaxf(m, v);
            }
            #pragma unroll
            for (int s = 1; s < 16; s <<= 1) m = fmaxf(m, __shfl_xor(m, s, 32));
            mrow[vg] = m;
        }
        if (l16 == 0) {
            #pragma unroll
            for (int vg = 0; vg < 8; ++vg) atomicMaxF(&rowmax[gr0 + vg], mrow[vg]);
        }
    }
}

// -------------------------------------------------------------------------
// Block reduction helper (wave32): returns total in (wid==0, lane==0)
// -------------------------------------------------------------------------
__device__ __forceinline__ float block_reduce_sum(float v, float* red8) {
    #pragma unroll
    for (int s = 16; s > 0; s >>= 1) v += __shfl_xor(v, s, 32);
    int lane = threadIdx.x & 31, wid = threadIdx.x >> 5;
    if (lane == 0) red8[wid] = v;
    __syncthreads();
    float t = 0.0f;
    if (wid == 0) {
        t = (lane < 8) ? red8[lane] : 0.0f;
        #pragma unroll
        for (int s = 4; s > 0; s >>= 1) t += __shfl_xor(t, s, 32);
    }
    return t;
}

// -------------------------------------------------------------------------
// K2: per-row sum of exp(l - rowmax) for both matrices. One block per row.
// -------------------------------------------------------------------------
__global__ __launch_bounds__(256) void sumexp_kernel(
        const unsigned short* __restrict__ ls, const unsigned short* __restrict__ lt,
        const float* __restrict__ max_s, const float* __restrict__ max_t,
        float* __restrict__ sum_s, float* __restrict__ sum_t) {
    __shared__ float red8[8];
    __shared__ float red8b[8];
    int row = blockIdx.x;
    float ms = max_s[row], mt = max_t[row];
    const unsigned int* prs = (const unsigned int*)(ls + (size_t)row * VOCAB);
    const unsigned int* prt = (const unsigned int*)(lt + (size_t)row * VOCAB);
    float accs = 0.0f, acct = 0.0f;
    for (int c = threadIdx.x; c < VOCAB / 2; c += 256) {
        unsigned int vs = prs[c], vt = prt[c];
        accs += __expf(bf2f(vs & 0xFFFFu) - ms) + __expf(bf2f(vs >> 16) - ms);
        acct += __expf(bf2f(vt & 0xFFFFu) - mt) + __expf(bf2f(vt >> 16) - mt);
    }
    float ts = block_reduce_sum(accs, red8);
    __syncthreads();
    float tt = block_reduce_sum(acct, red8b);
    if (threadIdx.x == 0) { sum_s[row] = ts; sum_t[row] = tt; }
}

// -------------------------------------------------------------------------
// K3: generalized JSD (beta = 0.5) reduction. One block per row.
// -------------------------------------------------------------------------
__device__ __forceinline__ float jsd_term(float s, float t, float lzs, float lzt) {
    float log_q = s - lzs;            // student log prob
    float log_p = t - lzt;            // teacher log prob
    float q = __expf(log_q);
    float p = __expf(log_p);
    float m = 0.5f * (p + q);
    float log_m = __logf(m);
    return 0.5f * (p * (log_p - log_m) + q * (log_q - log_m));
}

__global__ __launch_bounds__(256) void jsd_kernel(
        const unsigned short* __restrict__ ls, const unsigned short* __restrict__ lt,
        const float* __restrict__ max_s, const float* __restrict__ max_t,
        const float* __restrict__ sum_s, const float* __restrict__ sum_t,
        float* __restrict__ out) {
    __shared__ float red8[8];
    int row = blockIdx.x;
    float lzs = max_s[row] + __logf(sum_s[row]);   // logZ student
    float lzt = max_t[row] + __logf(sum_t[row]);   // logZ teacher
    const unsigned int* prs = (const unsigned int*)(ls + (size_t)row * VOCAB);
    const unsigned int* prt = (const unsigned int*)(lt + (size_t)row * VOCAB);
    float acc = 0.0f;
    for (int c = threadIdx.x; c < VOCAB / 2; c += 256) {
        unsigned int vs = prs[c], vt = prt[c];
        acc += jsd_term(bf2f(vs & 0xFFFFu), bf2f(vt & 0xFFFFu), lzs, lzt);
        acc += jsd_term(bf2f(vs >> 16), bf2f(vt >> 16), lzs, lzt);
    }
    float total = block_reduce_sum(acc, red8);
    if (threadIdx.x == 0) atomicAdd(out, total * (1.0f / (float)BT));
}

// -------------------------------------------------------------------------
extern "C" void kernel_launch(void* const* d_in, const int* in_sizes, int n_in,
                              void* d_out, int out_size, void* d_ws, size_t ws_size,
                              hipStream_t stream) {
    const float* sx = (const float*)d_in[0];   // student_input  [BT, H]
    const float* sw = (const float*)d_in[1];   // student_weight [V, H]
    const float* tx = (const float*)d_in[2];   // teacher_input  [BT, H]
    const float* tw = (const float*)d_in[3];   // teacher_weight [V, H]
    float* out = (float*)d_out;

    char* ws = (char*)d_ws;
    size_t logitsBytes = (size_t)BT * VOCAB * sizeof(unsigned short);   // 131 MB
    size_t xBytes = (size_t)BT * HDIM * sizeof(unsigned short);         // 8.4 MB
    size_t wBytes = (size_t)VOCAB * HDIM * sizeof(unsigned short);      // 131 MB

    unsigned short* ls = (unsigned short*)ws;
    unsigned short* lt = (unsigned short*)(ws + logitsBytes);
    unsigned short* xs = (unsigned short*)(ws + 2 * logitsBytes);
    unsigned short* wsb = (unsigned short*)(ws + 2 * logitsBytes + xBytes);
    unsigned short* xt = (unsigned short*)(ws + 2 * logitsBytes + xBytes + wBytes);
    unsigned short* wtb = (unsigned short*)(ws + 2 * logitsBytes + 2 * xBytes + wBytes);
    float* max_s = (float*)(ws + 2 * logitsBytes + 2 * xBytes + 2 * wBytes);
    float* max_t = max_s + BT;
    float* sum_s = max_t + BT;
    float* sum_t = sum_s + BT;

    init_kernel<<<(BT + 255) / 256, 256, 0, stream>>>(max_s, max_t, out);

    // One-time fp32 -> bf16 conversion (memory bound).
    int nx4 = BT * HDIM / 4;
    int nw4 = VOCAB * HDIM / 4;
    cvt_bf16_kernel<<<1024, 256, 0, stream>>>(sx, xs, nx4);
    cvt_bf16_kernel<<<4096, 256, 0, stream>>>(sw, wsb, nw4);
    cvt_bf16_kernel<<<1024, 256, 0, stream>>>(tx, xt, nx4);
    cvt_bf16_kernel<<<4096, 256, 0, stream>>>(tw, wtb, nw4);

    dim3 g1(VOCAB / TILE_N, BT / TILE_M);   // (250, 16)
    gemm_logits_kernel<<<g1, 256, 0, stream>>>(xs, wsb, ls, max_s);
    gemm_logits_kernel<<<g1, 256, 0, stream>>>(xt, wtb, lt, max_t);

    sumexp_kernel<<<BT, 256, 0, stream>>>(ls, lt, max_s, max_t, sum_s, sum_t);
    jsd_kernel<<<BT, 256, 0, stream>>>(ls, lt, max_s, max_t, sum_s, sum_t, out);
}
